// TrajectoryGenerator_10127532883978
// MI455X (gfx1250) — matrix operand order repaired and checked by reading, not compiled, not run on hardware
//
#include <hip/hip_runtime.h>
#include <math.h>

#define OBS_LEN 8
#define PRED_LEN 12
#define S_SCENES 512
#define NPED 64
#define NTOT (S_SCENES * NPED)   // 32768
#define EPSN 1e-5f

typedef __bf16 bf16;
typedef __attribute__((ext_vector_type(16))) __bf16 v16bf;
typedef __attribute__((ext_vector_type(8)))  float  v8f;
typedef __attribute__((ext_vector_type(4)))  unsigned int u32x4;
typedef __attribute__((ext_vector_type(8)))  int i32x8;
typedef __attribute__((ext_vector_type(4)))  int i32x4;

// ---------------------------------------------------------------------------
// WMMA helpers (CDNA5 wave32, D = A[16x32 bf16] * B[32x16 bf16] + C[16x16 f32])
// ---------------------------------------------------------------------------
__device__ __forceinline__ v8f wmma_bf16(v16bf a, v16bf b, v8f c) {
  return __builtin_amdgcn_wmma_f32_16x16x32_bf16(false, a, false, b, (short)0, c,
                                                 false, false);
}

// A fragment from row-major LDS tile. Lane L: M = L&15; element e -> K =
// k0 + 16*(L>>4) + e.
__device__ __forceinline__ v16bf load_a_frag(const bf16* base, int ldk, int k0) {
  const int lane = threadIdx.x & 31;
  const int m = lane & 15;
  const int kb = (lane >> 4) * 16;
  const bf16* p = base + m * ldk + k0 + kb;
  v16bf a;
#pragma unroll
  for (int e = 0; e < 16; ++e) a[e] = p[e];
  return a;
}

// B fragment from column-major ("Bt[col][k]") LDS staging. Lane L: N = L&15;
// element e -> K = k0 + 16*(L>>4) + e.
__device__ __forceinline__ v16bf load_b_frag(const bf16* bt, int col0, int ldk, int k0) {
  const int lane = threadIdx.x & 31;
  const int n = lane & 15;
  const bf16* p = bt + (col0 + n) * ldk + k0 + (lane >> 4) * 16;
  v16bf b;
#pragma unroll
  for (int e = 0; e < 16; ++e) b[e] = p[e];
  return b;
}

__device__ __forceinline__ float sigmoidf_(float x) { return 1.f / (1.f + __expf(-x)); }

// ---------------------------------------------------------------------------
// Tensor Data Mover: 1-D contiguous copy of nelem bf16 (2-byte) elements from
// global memory into LDS.  D# per CDNA5 ISA:
// group0: count=1, lds_addr, global_addr[56:0], type=2 ("image").
// group1: data_size=1 (2B), tensor_dim0=nelem, tensor_dim1=1, tile_dim0=nelem,
//         tile_dim1=0 (unused -> 1-D), stride0=nelem.  Groups 2/3 zero (<=2D).
// Tracked by TENSORcnt; wave-level op (EXEC ignored).
// This toolchain's builtin takes 6 args (extra i32x8 before cpol).
// ---------------------------------------------------------------------------
__device__ __forceinline__ void tdm_copy_1d(const void* lds_dst, const void* gsrc,
                                            unsigned nelem) {
  const unsigned lds_addr = (unsigned)(unsigned long long)lds_dst;
  const unsigned long long ga = (unsigned long long)gsrc;
  u32x4 g0 = {1u,                                   // count=1, user mode
              lds_addr,                             // lds_addr
              (unsigned)ga,                         // global_addr[31:0]
              ((unsigned)(ga >> 32) & 0x01FFFFFFu) | 0x80000000u}; // [56:32]|type=2
  i32x8 g1 = {(int)0x00010000u,                     // data_size=1 (2 bytes)
              (int)((nelem & 0xFFFFu) << 16),       // tensor_dim0[15:0] @ [63:48]
              (int)((nelem >> 16) | (1u << 16)),    // tensor_dim0[31:16], tensor_dim1=1
              (int)((nelem & 0xFFFFu) << 16),       // tile_dim0 @ [127:112]
              0,                                    // tile_dim1=0, tile_dim2=0
              (int)nelem,                           // tensor_dim0_stride[31:0]
              0, 0};
  i32x4 gz4 = {0, 0, 0, 0};
  i32x8 gz8 = {0, 0, 0, 0, 0, 0, 0, 0};
  __builtin_amdgcn_tensor_load_to_lds(g0, g1, gz4, gz4, gz8, 0);
}

// ===========================================================================
// Kernel 0: build bf16 weight images (WMMA B-operand layouts) in workspace.
//   [0]      pWhT_traj [128][32]
//   [4096]   pWiT_g    [128][32]
//   [8192]   pWhT_g    [128][32]
//   [12288]  pBtw0     [64][32]   (col=h*16+o, k)
//   [14336]  pBtw1     [32][64]   (col=o, k)
//   [16384]  pWpT      [320][96]  (K zero-padded 80 -> 96)
// ===========================================================================
__global__ __launch_bounds__(256) void prep_weights_kernel(
    const float* __restrict__ twhh, const float* __restrict__ gwih,
    const float* __restrict__ gwhh, const float* __restrict__ w0,
    const float* __restrict__ w1, const float* __restrict__ pwhh,
    bf16* __restrict__ out)
{
  const int tid = threadIdx.x;
  bf16* pT  = out;
  bf16* pGi = out + 4096;
  bf16* pGh = out + 8192;
  bf16* pB0 = out + 12288;
  bf16* pB1 = out + 14336;
  bf16* pP  = out + 16384;
  for (int i = tid; i < 4096; i += 256) {
    pT[i]  = (bf16)twhh[i];
    pGi[i] = (bf16)gwih[i];
    pGh[i] = (bf16)gwhh[i];
  }
  for (int i = tid; i < 2048; i += 256) {
    const int col = i >> 5, k = i & 31;
    pB0[col * 32 + k] = (bf16)w0[((col >> 4) * 32 + k) * 16 + (col & 15)];
    pB1[(i & 31) * 64 + (i >> 5)] = (bf16)w1[(i >> 5) * 32 + (i & 31)];
  }
  for (int i = tid; i < 320 * 96; i += 256) {
    const int c = i / 96, k = i % 96;
    pP[i] = (k < 80) ? (bf16)pwhh[c * 80 + k] : (bf16)0.f;
  }
}

// ===========================================================================
// Kernel 1: observation-encoder LSTM (2 -> 32), 8 steps.
// One wave = 16 peds. Gates[N,128] = x@Wih^T + h@Whh^T + b via WMMA.
// c-state lives in registers in WMMA-C layout; h round-trips via LDS (bf16).
// Whh image DMA'd into LDS by the TDM.
// ===========================================================================
__global__ __launch_bounds__(256) void traj_lstm_kernel(
    const float* __restrict__ obs,   // [8][N][2]
    const float* __restrict__ h0, const float* __restrict__ c0,
    const float* __restrict__ wih,   // [128][2]
    const bf16*  __restrict__ pWhT,  // [128][32] bf16 image
    const float* __restrict__ bih, const float* __restrict__ bhh,
    bf16* __restrict__ traj_hs)      // [8][N][32]
{
  __shared__ bf16  WhT[128 * 32];
  __shared__ float wih_s[128 * 2];
  __shared__ float bsum[128];
  __shared__ bf16  hbuf[8][16 * 32];
  __shared__ float xw[8][16 * 2];

  const int tid = threadIdx.x, w = tid >> 5, lane = tid & 31;
  const int hi = lane >> 4, nn = lane & 15;
  const int ped0 = blockIdx.x * 128 + w * 16;

  if (w == 0) tdm_copy_1d(WhT, pWhT, 128 * 32);
  wih_s[tid] = wih[tid];  // 256 entries, 256 threads
  if (tid < 128) bsum[tid] = bih[tid] + bhh[tid];
  if (lane < 16)
    for (int k = 0; k < 32; ++k) hbuf[w][lane * 32 + k] = (bf16)h0[(ped0 + lane) * 32 + k];

  float cst[2][8];
#pragma unroll
  for (int half = 0; half < 2; ++half)
#pragma unroll
    for (int r = 0; r < 8; ++r)
      cst[half][r] = c0[(ped0 + r + 8 * hi) * 32 + nn + 16 * half];
  __builtin_amdgcn_s_wait_tensorcnt(0);
  __syncthreads();

  for (int t = 0; t < OBS_LEN; ++t) {
    if (lane < 16) {
      xw[w][lane * 2 + 0] = obs[((size_t)t * NTOT + ped0 + lane) * 2 + 0];
      xw[w][lane * 2 + 1] = obs[((size_t)t * NTOT + ped0 + lane) * 2 + 1];
    }
    __syncthreads();
    float xr0[8], xr1[8];
#pragma unroll
    for (int r = 0; r < 8; ++r) {
      xr0[r] = xw[w][(r + 8 * hi) * 2];
      xr1[r] = xw[w][(r + 8 * hi) * 2 + 1];
    }
    v16bf a = load_a_frag(hbuf[w], 32, 0);
    v8f acc[8];
#pragma unroll
    for (int ct = 0; ct < 8; ++ct) {
      const int col = ct * 16 + nn;
      const float bb = bsum[col], w0 = wih_s[col * 2], w1 = wih_s[col * 2 + 1];
#pragma unroll
      for (int r = 0; r < 8; ++r) acc[ct][r] = bb + xr0[r] * w0 + xr1[r] * w1;
      acc[ct] = wmma_bf16(a, load_b_frag(WhT, ct * 16, 32, 0), acc[ct]);
    }
    __syncthreads();
#pragma unroll
    for (int half = 0; half < 2; ++half)
#pragma unroll
      for (int r = 0; r < 8; ++r) {
        float ig = sigmoidf_(acc[0 + half][r]);
        float fg = sigmoidf_(acc[2 + half][r]);
        float gg = tanhf(acc[4 + half][r]);
        float og = sigmoidf_(acc[6 + half][r]);
        float cv = fg * cst[half][r] + ig * gg;
        cst[half][r] = cv;
        float hv = og * tanhf(cv);
        const int m = r + 8 * hi, k = nn + 16 * half;
        hbuf[w][m * 32 + k] = (bf16)hv;
        traj_hs[((size_t)t * NTOT + ped0 + m) * 32 + k] = (bf16)hv;
      }
    __syncthreads();
  }
}

// ===========================================================================
// Kernel 2: GAT encoder, one block per (scene,t): inorm -> GAT0(4 heads) ->
// elu -> inorm -> GAT1 -> graph_in. All GEMMs via WMMA, softmax in f32.
// x tile + both weight images arrive via TDM.
// ===========================================================================
__global__ __launch_bounds__(256) void gat_kernel(
    const bf16* __restrict__ traj_hs,  // [8][N][32]
    const bf16* __restrict__ pB0,      // [64][32] bf16 image
    const float* __restrict__ as0, const float* __restrict__ ad0,
    const float* __restrict__ b0,      // [16]
    const bf16* __restrict__ pB1,      // [32][64] bf16 image
    const float* __restrict__ as1, const float* __restrict__ ad1,
    const float* __restrict__ b1,      // [32]
    bf16* __restrict__ graph_in)       // [8][N][32]
{
  __shared__ float xn[64 * 64];     // layer input (f32), pitch 64
  __shared__ bf16  xbf[64 * 64];    // A staging
  __shared__ float hp[64 * 64];     // projected features (f32)
  __shared__ bf16  hpT[64 * 64];    // Bt staging of hp (col-major)
  __shared__ bf16  attnbf[64 * 64]; // one head's softmax weights (A operand)
  __shared__ bf16  xg[64 * 32];     // TDM landing pad for x tile
  __shared__ bf16  wst0[64 * 32];   // layer-0 weight image
  __shared__ bf16  wst1[32 * 64];   // layer-1 weight image
  __shared__ float asc[4 * 64], adc[4 * 64];
  __shared__ float avec[64], dvec[64], bvec[32];
  __shared__ float avec1[32], dvec1[32], bvec1[32];
  __shared__ float mu[64], iv[64];

  const int tid = threadIdx.x, w = tid >> 5, lane = tid & 31;
  const int hi = lane >> 4, nn = lane & 15;
  const int t = blockIdx.x & 7, s = blockIdx.x >> 3;
  const int pbase = s * NPED;

  // -------- async: x tile + both weight stages --------
  if (w == 0) {
    tdm_copy_1d(xg, traj_hs + ((size_t)t * NTOT + pbase) * 32, 64 * 32);
    tdm_copy_1d(wst0, pB0, 64 * 32);
    tdm_copy_1d(wst1, pB1, 32 * 64);
  }
  if (tid < 64) { avec[tid] = as0[tid]; dvec[tid] = ad0[tid]; }
  if (tid < 16) bvec[tid] = b0[tid];
  if (tid < 32) { avec1[tid] = as1[tid]; dvec1[tid] = ad1[tid]; bvec1[tid] = b1[tid]; }
  __builtin_amdgcn_s_wait_tensorcnt(0);
  __syncthreads();
  for (int i = tid; i < 64 * 32; i += 256) {
    const int n = i >> 5, c = i & 31;
    xn[n * 64 + c] = (float)xg[i];
  }
  __syncthreads();

  // -------- instance norm over peds (32 channels) --------
  if (tid < 32) {
    float s1 = 0.f, s2 = 0.f;
    for (int n = 0; n < 64; ++n) { float v = xn[n * 64 + tid]; s1 += v; s2 += v * v; }
    float m_ = s1 * (1.f / 64.f), v_ = s2 * (1.f / 64.f) - m_ * m_;
    mu[tid] = m_; iv[tid] = rsqrtf(v_ + EPSN);
  }
  __syncthreads();
  for (int i = tid; i < 64 * 32; i += 256) {
    const int n = i >> 5, c = i & 31;
    float v = (xn[n * 64 + c] - mu[c]) * iv[c];
    xn[n * 64 + c] = v;
    xbf[n * 64 + c] = (bf16)v;
  }
  __syncthreads();

  // -------- hp = xnorm @ w0 : M=64, Ncol=64, K=32 (16 WMMA jobs) --------
  for (int job = w; job < 16; job += 8) {
    const int mt = job >> 2, ct = job & 3;
    v16bf a = load_a_frag(xbf + mt * 16 * 64, 64, 0);
    v16bf b = load_b_frag(wst0, ct * 16, 32, 0);
    v8f acc = {};
    acc = wmma_bf16(a, b, acc);
    const int col = ct * 16 + nn;
#pragma unroll
    for (int r = 0; r < 8; ++r) {
      const int row = mt * 16 + r + 8 * hi;
      hp[row * 64 + col] = acc[r];
      hpT[col * 64 + row] = (bf16)acc[r];
    }
  }
  __syncthreads();

  // -------- additive attention scores --------
  {
    const int h = tid >> 6, n = tid & 63;
    float sa = 0.f, sd = 0.f;
    for (int o = 0; o < 16; ++o) {
      float v = hp[n * 64 + h * 16 + o];
      sa += v * avec[h * 16 + o];
      sd += v * dvec[h * 16 + o];
    }
    asc[h * 64 + n] = sa; adc[h * 64 + n] = sd;
  }
  __syncthreads();

  // -------- per head: softmax row-wise then out = attn @ hp + b0, elu --------
  for (int h = 0; h < 4; ++h) {
    if (tid < 64) {
      const int n = tid;
      const float an = asc[h * 64 + n];
      float mx = -1e30f;
      for (int m = 0; m < 64; ++m) {
        float e = an + adc[h * 64 + m]; e = e >= 0.f ? e : 0.2f * e;
        mx = fmaxf(mx, e);
      }
      float sm = 0.f;
      for (int m = 0; m < 64; ++m) {
        float e = an + adc[h * 64 + m]; e = e >= 0.f ? e : 0.2f * e;
        sm += __expf(e - mx);
      }
      const float inv = 1.f / sm;
      for (int m = 0; m < 64; ++m) {
        float e = an + adc[h * 64 + m]; e = e >= 0.f ? e : 0.2f * e;
        attnbf[n * 64 + m] = (bf16)(__expf(e - mx) * inv);
      }
    }
    __syncthreads();
    if (w < 4) {   // wave-uniform; 4 waves, one M-tile each
      const int mt = w;
      v8f acc;
      const float bb = bvec[nn];
#pragma unroll
      for (int r = 0; r < 8; ++r) acc[r] = bb;
#pragma unroll
      for (int ks = 0; ks < 2; ++ks) {
        v16bf a = load_a_frag(attnbf + mt * 16 * 64, 64, ks * 32);
        v16bf b = load_b_frag(hpT, h * 16, 64, ks * 32);
        acc = wmma_bf16(a, b, acc);
      }
      const int col = h * 16 + nn;
#pragma unroll
      for (int r = 0; r < 8; ++r) {
        const int row = mt * 16 + r + 8 * hi;
        float v = acc[r];
        v = v > 0.f ? v : (__expf(v) - 1.f);  // elu
        xn[row * 64 + col] = v;
      }
    }
    __syncthreads();
  }

  // -------- instance norm (64 channels) --------
  if (tid < 64) {
    float s1 = 0.f, s2 = 0.f;
    for (int n = 0; n < 64; ++n) { float v = xn[n * 64 + tid]; s1 += v; s2 += v * v; }
    float m_ = s1 * (1.f / 64.f), v_ = s2 * (1.f / 64.f) - m_ * m_;
    mu[tid] = m_; iv[tid] = rsqrtf(v_ + EPSN);
  }
  __syncthreads();
  for (int i = tid; i < 64 * 64; i += 256) {
    const int n = i >> 6, c = i & 63;
    float v = (xn[n * 64 + c] - mu[c]) * iv[c];
    xn[n * 64 + c] = v; xbf[n * 64 + c] = (bf16)v;
  }
  __syncthreads();

  // -------- hp1 = xn @ w1 : M=64, Ncol=32, K=64 (8 jobs) --------
  {
    const int mt = w >> 1, ct = w & 1;
    v8f acc = {};
#pragma unroll
    for (int ks = 0; ks < 2; ++ks) {
      v16bf a = load_a_frag(xbf + mt * 16 * 64, 64, ks * 32);
      v16bf b = load_b_frag(wst1, ct * 16, 64, ks * 32);
      acc = wmma_bf16(a, b, acc);
    }
    const int col = ct * 16 + nn;
#pragma unroll
    for (int r = 0; r < 8; ++r) {
      const int row = mt * 16 + r + 8 * hi;
      hp[row * 64 + col] = acc[r];
      hpT[col * 64 + row] = (bf16)acc[r];
    }
  }
  __syncthreads();

  if (tid < 64) {
    float sa = 0.f, sd = 0.f;
    for (int o = 0; o < 32; ++o) { float v = hp[tid * 64 + o]; sa += v * avec1[o]; sd += v * dvec1[o]; }
    asc[tid] = sa; adc[tid] = sd;
  }
  __syncthreads();
  if (tid < 64) {
    const float an = asc[tid];
    float mx = -1e30f;
    for (int m = 0; m < 64; ++m) { float e = an + adc[m]; e = e >= 0.f ? e : 0.2f * e; mx = fmaxf(mx, e); }
    float sm = 0.f;
    for (int m = 0; m < 64; ++m) { float e = an + adc[m]; e = e >= 0.f ? e : 0.2f * e; sm += __expf(e - mx); }
    const float inv = 1.f / sm;
    for (int m = 0; m < 64; ++m) {
      float e = an + adc[m]; e = e >= 0.f ? e : 0.2f * e;
      attnbf[tid * 64 + m] = (bf16)(__expf(e - mx) * inv);
    }
  }
  __syncthreads();

  // -------- out1 = attn1 @ hp1 + b1 -> graph_in --------
  {
    const int mt = w >> 1, ct = w & 1;
    const int col = ct * 16 + nn;
    v8f acc;
    const float bb = bvec1[col];
#pragma unroll
    for (int r = 0; r < 8; ++r) acc[r] = bb;
#pragma unroll
    for (int ks = 0; ks < 2; ++ks) {
      v16bf a = load_a_frag(attnbf + mt * 16 * 64, 64, ks * 32);
      v16bf b = load_b_frag(hpT, ct * 16, 64, ks * 32);
      acc = wmma_bf16(a, b, acc);
    }
#pragma unroll
    for (int r = 0; r < 8; ++r) {
      const int row = mt * 16 + r + 8 * hi;
      graph_in[((size_t)t * NTOT + pbase + row) * 32 + col] = (bf16)acc[r];
    }
  }
}

// ===========================================================================
// Kernel 3: graph LSTM (32 -> 32) with LayerNorm, 8 steps.
// Weight images via TDM; per-step x tiles TDM double-buffered per wave
// (issue t+1 during step t, s_wait_tensorcnt 1 before consuming t).
// ===========================================================================
__global__ __launch_bounds__(256) void graph_lstm_kernel(
    const bf16* __restrict__ graph_in,  // [8][N][32]
    const float* __restrict__ h0, const float* __restrict__ c0,
    const bf16* __restrict__ pWiT,      // [128][32] bf16 image
    const bf16* __restrict__ pWhT,      // [128][32] bf16 image
    const float* __restrict__ bih, const float* __restrict__ bhh,
    const float* __restrict__ lng, const float* __restrict__ lnb,
    float* __restrict__ graph_h)        // [N][32]
{
  __shared__ bf16  WiT[128 * 32];
  __shared__ bf16  WhT[128 * 32];
  __shared__ float bsum[128];
  __shared__ float lg[32], lb[32];
  __shared__ bf16  hbuf[8][16 * 32];
  __shared__ bf16  xbuf[8][2][16 * 32];  // double-buffered TDM landing pads

  const int tid = threadIdx.x, w = tid >> 5, lane = tid & 31;
  const int hi = lane >> 4, nn = lane & 15;
  const int ped0 = blockIdx.x * 128 + w * 16;

  if (w == 0) {
    tdm_copy_1d(WiT, pWiT, 128 * 32);
    tdm_copy_1d(WhT, pWhT, 128 * 32);
  }
  tdm_copy_1d(xbuf[w][0], graph_in + (size_t)ped0 * 32, 16 * 32);  // tile t=0

  if (tid < 128) bsum[tid] = bih[tid] + bhh[tid];
  if (tid < 32) { lg[tid] = lng[tid]; lb[tid] = lnb[tid]; }
  if (lane < 16)
    for (int k = 0; k < 32; ++k) hbuf[w][lane * 32 + k] = (bf16)h0[(ped0 + lane) * 32 + k];
  float cst[2][8];
#pragma unroll
  for (int half = 0; half < 2; ++half)
#pragma unroll
    for (int r = 0; r < 8; ++r)
      cst[half][r] = c0[(ped0 + r + 8 * hi) * 32 + nn + 16 * half];
  __builtin_amdgcn_s_wait_tensorcnt(0);
  __syncthreads();

  for (int t = 0; t < OBS_LEN; ++t) {
    if (t < OBS_LEN - 1) {
      tdm_copy_1d(xbuf[w][(t + 1) & 1],
                  graph_in + ((size_t)(t + 1) * NTOT + ped0) * 32, 16 * 32);
      __builtin_amdgcn_s_wait_tensorcnt(1);  // tile t landed (in-order)
    } else {
      __builtin_amdgcn_s_wait_tensorcnt(0);
    }
    v16bf ax = load_a_frag(xbuf[w][t & 1], 32, 0);
    v16bf ah = load_a_frag(hbuf[w], 32, 0);
    v8f acc[8];
#pragma unroll
    for (int ct = 0; ct < 8; ++ct) {
      const float bb = bsum[ct * 16 + nn];
#pragma unroll
      for (int r = 0; r < 8; ++r) acc[ct][r] = bb;
      acc[ct] = wmma_bf16(ax, load_b_frag(WiT, ct * 16, 32, 0), acc[ct]);
      acc[ct] = wmma_bf16(ah, load_b_frag(WhT, ct * 16, 32, 0), acc[ct]);
    }
    __syncthreads();
#pragma unroll
    for (int half = 0; half < 2; ++half)
#pragma unroll
      for (int r = 0; r < 8; ++r) {
        float ig = sigmoidf_(acc[0 + half][r]);
        float fg = sigmoidf_(acc[2 + half][r]);
        float gg = tanhf(acc[4 + half][r]);
        float og = sigmoidf_(acc[6 + half][r]);
        float cv = fg * cst[half][r] + ig * gg;
        cst[half][r] = cv;
        hbuf[w][(r + 8 * hi) * 32 + nn + 16 * half] = (bf16)(og * tanhf(cv));
      }
    __syncthreads();
    if (lane < 16) {  // layer-norm per ped, in place on hbuf
      float s1 = 0.f, s2 = 0.f;
      for (int k = 0; k < 32; ++k) { float v = (float)hbuf[w][lane * 32 + k]; s1 += v; s2 += v * v; }
      float m_ = s1 * (1.f / 32.f), v_ = s2 * (1.f / 32.f) - m_ * m_;
      float ivv = rsqrtf(v_ + EPSN);
      for (int k = 0; k < 32; ++k) {
        float v = ((float)hbuf[w][lane * 32 + k] - m_) * ivv * lg[k] + lb[k];
        hbuf[w][lane * 32 + k] = (bf16)v;
        if (t == OBS_LEN - 1) graph_h[(ped0 + lane) * 32 + k] = v;
      }
    }
    __syncthreads();
  }
}

// ===========================================================================
// Kernel 4: autoregressive decoder LSTM (2 -> 80), 12 steps, LN + 2-d head.
// K=80 zero-padded to 96 -> 3 WMMA k-steps; 20 column tiles processed in 5
// groups so (i,f,g,o) for the same hidden range coexist in registers.
// 60 KB padded Whh image arrives via a single TDM descriptor.
// ===========================================================================
__global__ __launch_bounds__(256) void decoder_kernel(
    const float* __restrict__ obs,      // [8][N][2]
    const bf16*  __restrict__ traj_hs,  // [8][N][32]
    const float* __restrict__ graph_h,  // [N][32]
    const float* __restrict__ z_noise,  // [512][16]
    const float* __restrict__ wih,      // [320][2]
    const bf16*  __restrict__ pWpT,     // [320][96] bf16 image (zero-padded)
    const float* __restrict__ bih, const float* __restrict__ bhh,
    const float* __restrict__ lng, const float* __restrict__ lnb,   // [80]
    const float* __restrict__ outw,     // [2][80]
    const float* __restrict__ outb,     // [2]
    float* __restrict__ action_mean)    // [12][N][2]
{
  __shared__ bf16  WpT[320 * 96];
  __shared__ float wih_s[320 * 2];
  __shared__ float bsum[320];
  __shared__ float lg[80], lb[80], ow[160], ob[2];
  __shared__ bf16  hstage[8][16 * 96];
  __shared__ float xw[8][16 * 2];

  const int tid = threadIdx.x, w = tid >> 5, lane = tid & 31;
  const int hi = lane >> 4, nn = lane & 15;
  const int ped0 = blockIdx.x * 128 + w * 16;

  if (w == 0) tdm_copy_1d(WpT, pWpT, 320 * 96);
  for (int i = tid; i < 640; i += 256) wih_s[i] = wih[i];
  for (int i = tid; i < 320; i += 256) bsum[i] = bih[i] + bhh[i];
  if (tid < 80) { lg[tid] = lng[tid]; lb[tid] = lnb[tid]; }
  if (tid < 160) ow[tid] = outw[tid];
  if (tid < 2) ob[tid] = outb[tid];

  if (lane < 16) {
    const int ped = ped0 + lane;
    for (int k = 0; k < 32; ++k) hstage[w][lane * 96 + k] = traj_hs[((size_t)7 * NTOT + ped) * 32 + k];
    for (int k = 0; k < 32; ++k) hstage[w][lane * 96 + 32 + k] = (bf16)graph_h[ped * 32 + k];
    for (int k = 0; k < 16; ++k) hstage[w][lane * 96 + 64 + k] = (bf16)z_noise[(ped >> 6) * 16 + k];
    for (int k = 80; k < 96; ++k) hstage[w][lane * 96 + k] = (bf16)0.f;
    xw[w][lane * 2 + 0] = obs[((size_t)7 * NTOT + ped) * 2 + 0];
    xw[w][lane * 2 + 1] = obs[((size_t)7 * NTOT + ped) * 2 + 1];
  }
  float cst[5][8];
#pragma unroll
  for (int ct = 0; ct < 5; ++ct)
#pragma unroll
    for (int r = 0; r < 8; ++r) cst[ct][r] = 0.f;
  __builtin_amdgcn_s_wait_tensorcnt(0);
  __syncthreads();

  for (int t = 0; t < PRED_LEN; ++t) {
    float xr0[8], xr1[8];
#pragma unroll
    for (int r = 0; r < 8; ++r) {
      xr0[r] = xw[w][(r + 8 * hi) * 2];
      xr1[r] = xw[w][(r + 8 * hi) * 2 + 1];
    }
    v16bf a0 = load_a_frag(hstage[w], 96, 0);
    v16bf a1 = load_a_frag(hstage[w], 96, 32);
    v16bf a2 = load_a_frag(hstage[w], 96, 64);

#pragma unroll
    for (int ct = 0; ct < 5; ++ct) {
      v8f acc[4];
#pragma unroll
      for (int gt = 0; gt < 4; ++gt) {
        const int col = gt * 80 + ct * 16 + nn;
        const float bb = bsum[col], w0 = wih_s[col * 2], w1 = wih_s[col * 2 + 1];
#pragma unroll
        for (int r = 0; r < 8; ++r) acc[gt][r] = bb + xr0[r] * w0 + xr1[r] * w1;
      }
#pragma unroll
      for (int gt = 0; gt < 4; ++gt) {
        const int col0 = gt * 80 + ct * 16;
        acc[gt] = wmma_bf16(a0, load_b_frag(WpT, col0, 96, 0),  acc[gt]);
        acc[gt] = wmma_bf16(a1, load_b_frag(WpT, col0, 96, 32), acc[gt]);
        acc[gt] = wmma_bf16(a2, load_b_frag(WpT, col0, 96, 64), acc[gt]);
      }
#pragma unroll
      for (int r = 0; r < 8; ++r) {
        float ig = sigmoidf_(acc[0][r]);
        float fg = sigmoidf_(acc[1][r]);
        float gg = tanhf(acc[2][r]);
        float og = sigmoidf_(acc[3][r]);
        float cv = fg * cst[ct][r] + ig * gg;
        cst[ct][r] = cv;
        hstage[w][(r + 8 * hi) * 96 + ct * 16 + nn] = (bf16)(og * tanhf(cv));
      }
    }
    __syncthreads();
    if (lane < 16) {  // LayerNorm + output head per ped
      const int m = lane, ped = ped0 + lane;
      float s1 = 0.f, s2 = 0.f;
      for (int k = 0; k < 80; ++k) { float v = (float)hstage[w][m * 96 + k]; s1 += v; s2 += v * v; }
      float mean = s1 * (1.f / 80.f), var = s2 * (1.f / 80.f) - mean * mean;
      float ivv = rsqrtf(var + EPSN);
      float o0 = 0.f, o1 = 0.f;
      for (int k = 0; k < 80; ++k) {
        float v = ((float)hstage[w][m * 96 + k] - mean) * ivv * lg[k] + lb[k];
        hstage[w][m * 96 + k] = (bf16)v;
        o0 += v * ow[k]; o1 += v * ow[80 + k];
      }
      o0 += ob[0]; o1 += ob[1];
      action_mean[((size_t)t * NTOT + ped) * 2 + 0] = o0;
      action_mean[((size_t)t * NTOT + ped) * 2 + 1] = o1;
      xw[w][m * 2] = o0; xw[w][m * 2 + 1] = o1;
    }
    __syncthreads();
  }
}

// ===========================================================================
// Kernel 5: fill log_std and std thirds of d_out.
// ===========================================================================
__global__ void fill_std_kernel(const float* __restrict__ als, float* __restrict__ out, int M) {
  const int i = blockIdx.x * blockDim.x + threadIdx.x;
  if (i < M) {
    const float v = als[i & 1];
    out[M + i] = v;
    out[2 * M + i] = __expf(v);
  }
}

// ===========================================================================
extern "C" void kernel_launch(void* const* d_in, const int* in_sizes, int n_in,
                              void* d_out, int out_size, void* d_ws, size_t ws_size,
                              hipStream_t stream) {
  (void)in_sizes; (void)n_in; (void)out_size; (void)ws_size;
  const float* obs      = (const float*)d_in[0];
  // d_in[1] = seq_start_end (uniform scenes -> unused)
  const float* z_noise  = (const float*)d_in[2];
  const float* traj_h0  = (const float*)d_in[3];
  const float* traj_c0  = (const float*)d_in[4];
  const float* graph_h0 = (const float*)d_in[5];
  const float* graph_c0 = (const float*)d_in[6];
  const float* gw0  = (const float*)d_in[7];
  const float* gas0 = (const float*)d_in[8];
  const float* gad0 = (const float*)d_in[9];
  const float* gb0  = (const float*)d_in[10];
  const float* gw1  = (const float*)d_in[11];
  const float* gas1 = (const float*)d_in[12];
  const float* gad1 = (const float*)d_in[13];
  const float* gb1  = (const float*)d_in[14];
  const float* twih = (const float*)d_in[15];
  const float* twhh = (const float*)d_in[16];
  const float* tbih = (const float*)d_in[17];
  const float* tbhh = (const float*)d_in[18];
  const float* gwih = (const float*)d_in[19];
  const float* gwhh = (const float*)d_in[20];
  const float* gbih = (const float*)d_in[21];
  const float* gbhh = (const float*)d_in[22];
  const float* glng = (const float*)d_in[23];
  const float* glnb = (const float*)d_in[24];
  const float* pwih = (const float*)d_in[25];
  const float* pwhh = (const float*)d_in[26];
  const float* pbih = (const float*)d_in[27];
  const float* pbhh = (const float*)d_in[28];
  const float* plng = (const float*)d_in[29];
  const float* plnb = (const float*)d_in[30];
  const float* poutw = (const float*)d_in[31];
  const float* poutb = (const float*)d_in[32];
  const float* als   = (const float*)d_in[33];

  char* ws = (char*)d_ws;
  bf16*  traj_hs  = (bf16*)ws;                                         // 16 MB
  bf16*  graph_in = (bf16*)(ws + (size_t)OBS_LEN * NTOT * 32 * 2);     // 16 MB
  float* graph_h  = (float*)(ws + (size_t)2 * OBS_LEN * NTOT * 32 * 2); // 4 MB
  bf16*  wimg     = (bf16*)(ws + (size_t)2 * OBS_LEN * NTOT * 32 * 2 +
                            (size_t)NTOT * 32 * 4);                     // ~92 KB
  const bf16* pWhT_traj = wimg;
  const bf16* pWiT_g    = wimg + 4096;
  const bf16* pWhT_g    = wimg + 8192;
  const bf16* pB0       = wimg + 12288;
  const bf16* pB1       = wimg + 14336;
  const bf16* pWpT      = wimg + 16384;
  float* amean = (float*)d_out;

  prep_weights_kernel<<<1, 256, 0, stream>>>(twhh, gwih, gwhh, gw0, gw1, pwhh, wimg);
  traj_lstm_kernel<<<NTOT / 128, 256, 0, stream>>>(obs, traj_h0, traj_c0, twih,
                                                   pWhT_traj, tbih, tbhh, traj_hs);
  gat_kernel<<<S_SCENES * OBS_LEN, 256, 0, stream>>>(traj_hs, pB0, gas0, gad0, gb0,
                                                     pB1, gas1, gad1, gb1, graph_in);
  graph_lstm_kernel<<<NTOT / 128, 256, 0, stream>>>(graph_in, graph_h0, graph_c0,
                                                    pWiT_g, pWhT_g, gbih, gbhh,
                                                    glng, glnb, graph_h);
  decoder_kernel<<<NTOT / 128, 256, 0, stream>>>(obs, traj_hs, graph_h, z_noise, pwih,
                                                 pWpT, pbih, pbhh, plng, plnb, poutw,
                                                 poutb, amean);
  const int M = PRED_LEN * NTOT * 2;
  fill_std_kernel<<<(M + 255) / 256, 256, 0, stream>>>(als, amean, M);
}
